// SiameseSiren_5033701671583
// MI455X (gfx1250) — matrix operand order repaired and checked by reading, not compiled, not run on hardware
//
#include <hip/hip_runtime.h>
#include <hip/hip_bf16.h>

typedef __attribute__((ext_vector_type(16))) _Float16 v16h;
typedef __attribute__((ext_vector_type(8)))  float    v8f;

#define MTILE      64
#define THREADS    256
#define ACT_STRIDE 258          // f16 elements per row (256 + 2 pad -> odd dword stride 129)
#define ACT_STR_DW (ACT_STRIDE/2)
#define KP2_MAX    128          // max K/2 pairs (K=256)

union Frag16 { unsigned int u[8]; v16h v; };

// Stage W (f32, [256 x Krow] row-major, computing x @ W.T) into LDS as packed f16x2
// in WMMA-B fragment order: wbuf[(nt*Kp2 + kpair)*16 + n] = pack(W[nt*16+n][2k], W[nt*16+n][2k+1])
__device__ __forceinline__ void load_weights(const float* __restrict__ W, int Krow, int Kp2,
                                             unsigned int* wbuf, int tid) {
    const int total = 256 * Kp2;
    for (int idx = tid; idx < total; idx += THREADS) {
        int row   = idx / Kp2;          // output neuron
        int kpair = idx - row * Kp2;    // consecutive threads -> consecutive k: coalesced
        int k     = kpair * 2;
        float w0 = (k     < Krow) ? W[row * Krow + k]     : 0.0f;
        float w1 = (k + 1 < Krow) ? W[row * Krow + k + 1] : 0.0f;
        union { _Float16 h[2]; unsigned int u; } p;
        p.h[0] = (_Float16)w0;
        p.h[1] = (_Float16)w1;
        int nt = row >> 4, n = row & 15;
        wbuf[(nt * Kp2 + kpair) * 16 + n] = p.u;
    }
}

// One dense layer: out[m][n] = sin(30 * (sum_k in[m][k]*W[n][k] + b[n])), 64x256 output tile.
// Each wave owns 2 n-tiles (processed one at a time to keep VGPR use < 256); for each it
// keeps 4 m-tile accumulators live so every B fragment is loaded from LDS once and feeds
// 4 back-to-back WMMAs (independent D registers -> no D->A/B hazard NOPs).
__device__ __forceinline__ void gemm_layer(const _Float16* __restrict__ actIn,
                                           _Float16* __restrict__ actOut,
                                           const unsigned int* __restrict__ wbuf,
                                           const float* __restrict__ bias,
                                           int nChunks, int Kp2, int wave, int lane) {
    const unsigned int* actInU = (const unsigned int*)actIn;
    const int h = lane >> 4;        // K-half select
    const int n = lane & 15;        // column within 16-wide tile / A row select
#pragma unroll 1
    for (int ntIdx = 0; ntIdx < 2; ++ntIdx) {
        const int nt = wave * 2 + ntIdx;
        v8f c[4] = {};
        for (int kb = 0; kb < nChunks; ++kb) {
            const int kbase = kb * 32;
            Frag16 b;
#pragma unroll
            for (int j = 0; j < 8; ++j) {
                // B fragment, contiguous k order (same as A): VGPR j holds K = 2j+16h, 2j+1+16h at column n
                b.u[j] = wbuf[(nt * Kp2 + (kbase >> 1) + j + 8 * h) * 16 + n];
            }
#pragma unroll
            for (int mt = 0; mt < 4; ++mt) {
                const int m = mt * 16 + n;      // A-matrix row for this lane
                Frag16 a;
#pragma unroll
                for (int j = 0; j < 8; ++j) {
                    // A fragment, contiguous k order (same as B): VGPR j holds K = 2j+16h, 2j+1+16h
                    const int koff = 2 * j + 16 * h;
                    a.u[j] = actInU[m * ACT_STR_DW + ((kbase + koff) >> 1)];
                }
                c[mt] = __builtin_amdgcn_wmma_f32_16x16x32_f16(
                            false, a.v, false, b.v, (short)0, c[mt], false, false);
            }
        }
        const float bv30 = 30.0f * bias[nt * 16 + n];
#pragma unroll
        for (int mt = 0; mt < 4; ++mt) {
#pragma unroll
            for (int r = 0; r < 8; ++r) {
                const int mo = mt * 16 + r + 8 * h;   // D layout: M = r + 8*half, N = lane&15
                float val = __sinf(fmaf(30.0f, c[mt][r], bv30));   // v_fma + v_mul + v_sin_f32
                actOut[mo * ACT_STRIDE + nt * 16 + n] = (_Float16)val;
            }
        }
    }
}

__global__ __launch_bounds__(THREADS, 1)
void siamese_siren_kernel(const float* __restrict__ coords,
                          const float* __restrict__ W0, const float* __restrict__ b0,
                          const float* __restrict__ W1, const float* __restrict__ b1,
                          const float* __restrict__ W2, const float* __restrict__ b2,
                          const float* __restrict__ Wl, const float* __restrict__ bl,
                          const float* __restrict__ Wr, const float* __restrict__ br,
                          const float* __restrict__ WfL, const float* __restrict__ bfL,
                          const float* __restrict__ WfR, const float* __restrict__ bfR,
                          float* __restrict__ out, int Ntot) {
    __shared__ _Float16 actA[MTILE * ACT_STRIDE];
    __shared__ _Float16 actB[MTILE * ACT_STRIDE];
    __shared__ unsigned int wbuf[16 * KP2_MAX * 16];   // 128KB weight staging

    const int tid  = threadIdx.x;
    const int wave = tid >> 5;
    const int lane = tid & 31;
    const int g0   = blockIdx.x * MTILE;

    // ---- positional encoding into actA (129 real cols, zero-padded to 160) ----
    for (int idx = tid; idx < MTILE * 160; idx += THREADS) {
        const int row = idx / 160;
        const int col = idx - row * 160;
        const float cv = coords[g0 + row];
        float v;
        if (col == 0) {
            v = cv;
        } else if (col < 129) {
            const int f = (col - 1) >> 1;
            const float freq = ldexpf(3.14159265358979323846f, f);   // pi * 2^f
            const float ang = cv * freq;
            v = ((col - 1) & 1) ? __cosf(ang) : __sinf(ang);
        } else {
            v = 0.0f;
        }
        actA[row * ACT_STRIDE + col] = (_Float16)v;
    }
    __syncthreads();

    // ---- layer 0: 160(pad of 129) -> 256 ----
    load_weights(W0, 129, 80, wbuf, tid);
    __syncthreads();
    gemm_layer(actA, actB, wbuf, b0, 5, 80, wave, lane);
    __syncthreads();

    // ---- layer 1 ----
    load_weights(W1, 256, 128, wbuf, tid);
    __syncthreads();
    gemm_layer(actB, actA, wbuf, b1, 8, 128, wave, lane);
    __syncthreads();

    // ---- layer 2 ----
    load_weights(W2, 256, 128, wbuf, tid);
    __syncthreads();
    gemm_layer(actA, actB, wbuf, b2, 8, 128, wave, lane);   // trunk h stays in actB
    __syncthreads();

    // ---- left branch ----
    load_weights(Wl, 256, 128, wbuf, tid);
    __syncthreads();
    gemm_layer(actB, actA, wbuf, bl, 8, 128, wave, lane);
    __syncthreads();
    // left head (256 -> 1); stage Wr weights concurrently (different LDS regions)
    load_weights(Wr, 256, 128, wbuf, tid);
    if (tid < MTILE) {
        float acc = bfL[0];
        for (int k = 0; k < 256; ++k)
            acc += (float)actA[tid * ACT_STRIDE + k] * WfL[k];
        out[(size_t)(g0 + tid) * 2 + 0] = acc;
    }
    __syncthreads();

    // ---- right branch (reuses actA) ----
    gemm_layer(actB, actA, wbuf, br, 8, 128, wave, lane);
    __syncthreads();
    if (tid < MTILE) {
        float acc = bfR[0];
        for (int k = 0; k < 256; ++k)
            acc += (float)actA[tid * ACT_STRIDE + k] * WfR[k];
        out[(size_t)(g0 + tid) * 2 + 1] = acc;
        // tuple second element: coords passthrough
        out[(size_t)2 * Ntot + g0 + tid] = coords[g0 + tid];
    }
}

extern "C" void kernel_launch(void* const* d_in, const int* in_sizes, int n_in,
                              void* d_out, int out_size, void* d_ws, size_t ws_size,
                              hipStream_t stream) {
    (void)n_in; (void)d_ws; (void)ws_size; (void)out_size;
    const float* coords = (const float*)d_in[0];
    const float* W0  = (const float*)d_in[1];
    const float* b0  = (const float*)d_in[2];
    const float* W1  = (const float*)d_in[3];
    const float* b1  = (const float*)d_in[4];
    const float* W2  = (const float*)d_in[5];
    const float* b2  = (const float*)d_in[6];
    const float* Wl  = (const float*)d_in[7];
    const float* bl  = (const float*)d_in[8];
    const float* Wr  = (const float*)d_in[9];
    const float* br  = (const float*)d_in[10];
    const float* WfL = (const float*)d_in[11];
    const float* bfL = (const float*)d_in[12];
    const float* WfR = (const float*)d_in[13];
    const float* bfR = (const float*)d_in[14];

    const int Ntot = in_sizes[0];          // coords is [N, 1]
    const int grid = Ntot / MTILE;         // 262144 / 64 = 4096

    siamese_siren_kernel<<<grid, THREADS, 0, stream>>>(
        coords, W0, b0, W1, b1, W2, b2, Wl, bl, Wr, br,
        WfL, bfL, WfR, bfR, (float*)d_out, Ntot);
}
